// HyenaEncoder_63299228009235
// MI455X (gfx1250) — matrix operand order repaired and checked by reading, not compiled, not run on hardware
//
#include <hip/hip_runtime.h>
#include <stdint.h>

// ---------------------------------------------------------------------------
// Types for CDNA5 WMMA (wave32)
// ---------------------------------------------------------------------------
typedef __attribute__((ext_vector_type(16))) __bf16 v16bf;
typedef __attribute__((ext_vector_type(8)))  float  v8f;
typedef __attribute__((ext_vector_type(4)))  float  f32x4;

union FragBF { v16bf v; f32x4 q[2]; };

struct cplx { float x, y; };

// f32 -> bf16 (round to nearest even), stored as raw u16 for LDS staging
static __device__ __forceinline__ unsigned short f2bf(float f) {
    union { float f; uint32_t u; } v; v.f = f;
    uint32_t r = v.u + 0x7FFFu + ((v.u >> 16) & 1u);
    return (unsigned short)(r >> 16);
}

static __device__ __forceinline__ float gelu_tanh(float x) {
    const float k0 = 0.7978845608028654f, k1 = 0.044715f;
    return 0.5f * x * (1.0f + tanhf(k0 * (x + k1 * x * x * x)));
}

// ---------------------------------------------------------------------------
// Embedding gather: u[b,l,:] = emb[ids[b,l],:]
// ---------------------------------------------------------------------------
__global__ void embed_kernel(const int* __restrict__ ids,
                             const float* __restrict__ emb,
                             float* __restrict__ u, int BL, int d) {
    size_t idx = (size_t)blockIdx.x * blockDim.x + threadIdx.x;
    size_t total = (size_t)BL * d;
    if (idx >= total) return;
    int row = (int)(idx / d), col = (int)(idx % d);
    u[idx] = emb[(size_t)ids[row] * d + col];
}

// ---------------------------------------------------------------------------
// LayerNorm: one wave (32 lanes) per row of d elements
// ---------------------------------------------------------------------------
__global__ __launch_bounds__(256) void ln_kernel(const float* __restrict__ in,
                                                 const float* __restrict__ g,
                                                 const float* __restrict__ bta,
                                                 float* __restrict__ out,
                                                 int rows, int d) {
    int wave = threadIdx.x >> 5, lane = threadIdx.x & 31;
    int row = blockIdx.x * 8 + wave;
    if (row >= rows) return;
    const float* x = in + (size_t)row * d;
    float s = 0.f, s2 = 0.f;
    for (int i = lane; i < d; i += 32) { float v = x[i]; s += v; s2 += v * v; }
    for (int off = 16; off; off >>= 1) {
        s  += __shfl_xor(s,  off, 32);
        s2 += __shfl_xor(s2, off, 32);
    }
    float mu = s / d;
    float var = s2 / d - mu * mu;
    float r = rsqrtf(var + 1e-5f);
    float* o = out + (size_t)row * d;
    for (int i = lane; i < d; i += 32) o[i] = (x[i] - mu) * r * g[i] + bta[i];
}

// ---------------------------------------------------------------------------
// bf16 WMMA GEMM: out[M,N] (= or +=) act( A[M,K] * W[K,N] + bias )
// Block tile 128x128 (256 threads = 8 wave32), wave tile 32x64 (2x4 WMMA),
// K step 32. 8 v_wmma per wave per K-iter, 6 fragment loads.
// flags bit0: accumulate into out; bit1: gelu
// M % 128 == 0, N % 128 == 0, K % 32 == 0 (true for all shapes here)
// ---------------------------------------------------------------------------
#define TM 128
#define TN 128
#define TK 32

__global__ __launch_bounds__(256) void gemm_wmma_kernel(
    const float* __restrict__ A, const float* __restrict__ W,
    const float* __restrict__ bias, float* __restrict__ out,
    int M, int N, int K, int flags) {
    __shared__ unsigned short As[TM * TK];  // [row][k]   8 KB
    __shared__ unsigned short Bs[TN * TK];  // [n][k]     8 KB (transposed)

    const int tid  = threadIdx.x;
    const int lane = tid & 31;
    const int wave = tid >> 5;
    const int m0 = blockIdx.y * TM;
    const int n0 = blockIdx.x * TN;
    const int wm = wave >> 1;   // 0..3 -> 32-row strip
    const int wn = wave & 1;    // 0..1 -> 64-col strip

    const v8f z8 = {0.f, 0.f, 0.f, 0.f, 0.f, 0.f, 0.f, 0.f};
    v8f acc[2][4];
#pragma unroll
    for (int i = 0; i < 2; ++i)
#pragma unroll
        for (int j = 0; j < 4; ++j) acc[i][j] = z8;

    const int hi = (lane >> 4) & 1;        // lane group 0..15 / 16..31
    const int arow_k = hi * 8;             // A frag K base
    const int bk     = hi * 8;             // B frag K base

    for (int k0 = 0; k0 < K; k0 += TK) {
        // stage A (f32 -> bf16): 128x32
        for (int i = tid; i < TM * TK; i += 256) {
            int r = i >> 5, c = i & 31;
            As[i] = f2bf(A[(size_t)(m0 + r) * K + (k0 + c)]);
        }
        // stage B transposed (coalesced global reads along N): 32x128
        for (int i = tid; i < TK * TN; i += 256) {
            int kk = i >> 7, n = i & 127;
            Bs[n * TK + kk] = f2bf(W[(size_t)(k0 + kk) * N + (n0 + n)]);
        }
        // prefetch next K tile into cache while this tile computes
        if (k0 + TK < K) {
            int r = tid & 127;
            __builtin_prefetch(&A[(size_t)(m0 + r) * K + (k0 + TK)], 0, 1);
            if (tid < TK) {
                __builtin_prefetch(&W[(size_t)(k0 + TK + tid) * N + n0], 0, 1);
                __builtin_prefetch(&W[(size_t)(k0 + TK + tid) * N + n0 + 64], 0, 1);
            }
        }
        __syncthreads();

        FragBF a[2], b[4];
#pragma unroll
        for (int i = 0; i < 2; ++i) {
            int row = wm * 32 + i * 16 + (lane & 15);
            const f32x4* p = (const f32x4*)&As[row * TK + arow_k];
            a[i].q[0] = p[0];   // K = kbase .. kbase+7
            a[i].q[1] = p[2];   // K = kbase+16 .. kbase+23
        }
#pragma unroll
        for (int j = 0; j < 4; ++j) {
            int col = wn * 64 + j * 16 + (lane & 15);
            const f32x4* p = (const f32x4*)&Bs[col * TK + bk];
            b[j].q[0] = p[0];   // K = bk .. bk+7
            b[j].q[1] = p[2];   // K = bk+16 .. bk+23
        }
#pragma unroll
        for (int i = 0; i < 2; ++i)
#pragma unroll
            for (int j = 0; j < 4; ++j)
                acc[i][j] = __builtin_amdgcn_wmma_f32_16x16x32_bf16(
                    false, a[i].v, false, b[j].v, (short)0, acc[i][j],
                    false, false);
        __syncthreads();
    }

    // writeback per C/D VGPR layout: VGPR r -> M = r (+8 for lanes 16..31), N = lane&15
#pragma unroll
    for (int i = 0; i < 2; ++i) {
#pragma unroll
        for (int j = 0; j < 4; ++j) {
            int col = n0 + wn * 64 + j * 16 + (lane & 15);
            float bv = bias ? bias[col] : 0.f;
#pragma unroll
            for (int r = 0; r < 8; ++r) {
                int row = m0 + wm * 32 + i * 16 + r + hi * 8;
                float v = acc[i][j][r] + bv;
                if (flags & 2) v = gelu_tanh(v);
                size_t idx = (size_t)row * N + col;
                out[idx] = (flags & 1) ? (out[idx] + v) : v;
            }
        }
    }
}

// ---------------------------------------------------------------------------
// Causal depthwise conv k=3 with transpose: reads p[B,L,C], writes q[B,C,L]
// out[b,c,l] = w0*p[l-2] + w1*p[l-1] + w2*p[l] + bias[c]
// ---------------------------------------------------------------------------
__global__ void shortconv_kernel(const float* __restrict__ p,
                                 const float* __restrict__ w,
                                 const float* __restrict__ bb,
                                 float* __restrict__ qout,
                                 int Bn, int C, int Ln) {
    size_t idx = (size_t)blockIdx.x * blockDim.x + threadIdx.x;
    size_t total = (size_t)Bn * C * Ln;
    if (idx >= total) return;
    int l = (int)(idx % Ln);
    size_t t = idx / Ln;
    int c = (int)(t % C);
    int b = (int)(t / C);
    const float* base = p + ((size_t)b * Ln) * C + c;
    float w0 = w[c * 3 + 0], w1 = w[c * 3 + 1], w2 = w[c * 3 + 2];
    float acc = w2 * base[(size_t)l * C];
    if (l >= 1) acc += w1 * base[(size_t)(l - 1) * C];
    if (l >= 2) acc += w0 * base[(size_t)(l - 2) * C];
    qout[idx] = acc + bb[c];
}

// ---------------------------------------------------------------------------
// In-LDS radix-2 FFT (Cooley-Tukey DIT with bit-reversal), complex f32
// ---------------------------------------------------------------------------
static __device__ void fft_inplace(cplx* Z, int NF, float sign, int tid, int nth) {
    const int lg = 31 - __clz(NF);
    for (int i = tid; i < NF; i += nth) {
        unsigned j = __brev((unsigned)i) >> (32 - lg);
        if (j > (unsigned)i) { cplx t = Z[i]; Z[i] = Z[j]; Z[j] = t; }
    }
    __syncthreads();
    for (int s = 1; s <= lg; ++s) {
        const int m = 1 << s, half = m >> 1;
        const float base = sign * 6.283185307179586f / (float)m;
        for (int t = tid; t < (NF >> 1); t += nth) {
            int pos = t & (half - 1);
            int grp = t >> (s - 1);
            int i1 = (grp << s) + pos;
            int i2 = i1 + half;
            float sn, cs;
            __sincosf(base * (float)pos, &sn, &cs);
            cplx u = Z[i1], w = Z[i2];
            float tx = cs * w.x - sn * w.y;
            float ty = cs * w.y + sn * w.x;
            Z[i1].x = u.x + tx; Z[i1].y = u.y + ty;
            Z[i2].x = u.x - tx; Z[i2].y = u.y - ty;
        }
        __syncthreads();
    }
}

// ---------------------------------------------------------------------------
// Fused Hyena fftconv: per (b, channel) workgroup, whole conv in 128KB LDS.
// z[n] = vg[n] + i*h[n] (zero-padded); one fwd FFT; Hermitian unpack ->
// Y = Vf*Hf; one inv FFT; y = (conv + Db*vg) * x2; output transposed [B,L,D].
// ---------------------------------------------------------------------------
__global__ __launch_bounds__(1024) void fftconv_kernel(
    const float* __restrict__ q,   // [B, 3D, L]  (x2 | x1 | v)
    const float* __restrict__ h,   // [D, L] layer filter
    const float* __restrict__ Db,  // [D]
    float* __restrict__ yt,        // [B, L, D]
    int Dch, int Ln) {
    extern __shared__ cplx Z[];    // 2*Ln complex
    const int NF = 2 * Ln;
    const int b = blockIdx.x / Dch;
    const int c = blockIdx.x % Dch;
    const float* x2 = q + ((size_t)b * 3 * Dch + c) * Ln;
    const float* x1 = q + ((size_t)b * 3 * Dch + Dch + c) * Ln;
    const float* vv = q + ((size_t)b * 3 * Dch + 2 * Dch + c) * Ln;
    const float* hr = h + (size_t)c * Ln;
    const int tid = threadIdx.x, nth = blockDim.x;

    for (int n = tid; n < Ln; n += nth) {
        Z[n].x = vv[n] * x1[n];   // gated input (real part)
        Z[n].y = hr[n];           // filter (imag part)
    }
    for (int n = Ln + tid; n < NF; n += nth) { Z[n].x = 0.f; Z[n].y = 0.f; }
    __syncthreads();

    fft_inplace(Z, NF, -1.0f, tid, nth);   // forward

    // Hermitian unpack of packed real signals, then pointwise product
    for (int k = tid; k <= (NF >> 1); k += nth) {
        int nk = (NF - k) & (NF - 1);
        cplx P = Z[k], Q = Z[nk];
        float Ax = 0.5f * (P.x + Q.x),  Ay = 0.5f * (P.y - Q.y);   // Vf
        float Bx = 0.5f * (P.y + Q.y),  By = -0.5f * (P.x - Q.x);  // Hf
        cplx Y;
        Y.x = Ax * Bx - Ay * By;
        Y.y = Ax * By + Ay * Bx;
        Z[k] = Y;
        Z[nk].x = Y.x; Z[nk].y = -Y.y;   // Hermitian symmetry
    }
    __syncthreads();

    fft_inplace(Z, NF, +1.0f, tid, nth);   // inverse (unscaled)

    const float inv = 1.0f / (float)NF;
    const float dbc = Db[c];
    for (int n = tid; n < Ln; n += nth) {
        float vg  = vv[n] * x1[n];
        float res = Z[n].x * inv;
        yt[((size_t)b * Ln + n) * Dch + c] = (res + dbc * vg) * x2[n];
    }
}

// ---------------------------------------------------------------------------
// Host driver
// ---------------------------------------------------------------------------
extern "C" void kernel_launch(void* const* d_in, const int* in_sizes, int n_in,
                              void* d_out, int out_size, void* d_ws, size_t ws_size,
                              hipStream_t stream) {
    (void)in_sizes; (void)n_in; (void)out_size; (void)ws_size;

    const int*   ids   = (const int*)  d_in[0];
    const float* emb   = (const float*)d_in[1];
    const float* ln1_g = (const float*)d_in[2];
    const float* ln1_b = (const float*)d_in[3];
    const float* Win   = (const float*)d_in[4];
    const float* bin_  = (const float*)d_in[5];
    const float* convw = (const float*)d_in[6];
    const float* convb = (const float*)d_in[7];
    const float* hfil  = (const float*)d_in[8];
    const float* Db    = (const float*)d_in[9];
    const float* Wout  = (const float*)d_in[10];
    const float* bout  = (const float*)d_in[11];
    const float* ln2_g = (const float*)d_in[12];
    const float* ln2_b = (const float*)d_in[13];
    const float* W1    = (const float*)d_in[14];
    const float* b1    = (const float*)d_in[15];
    const float* W2    = (const float*)d_in[16];
    const float* b2    = (const float*)d_in[17];
    const float* lnf_g = (const float*)d_in[18];
    const float* lnf_b = (const float*)d_in[19];

    constexpr int Bn = 2, Ln = 8192, Dd = 512, NL = 8;
    constexpr int D3 = 3 * Dd, D4 = 4 * Dd;
    constexpr int MT = Bn * Ln;                 // 16384 token rows
    constexpr size_t SZ_U = (size_t)MT * Dd * sizeof(float);      // 32 MB

    char* ws = (char*)d_ws;
    float* u  = (float*)ws;                                        // [B,L,D]
    float* zb = (float*)(ws + SZ_U);                               // [B,L,D] (z / y)
    float* pb = (float*)(ws + 2 * SZ_U);                           // [B,L,3D]
    float* qb = (float*)(ws + 2 * SZ_U + (size_t)MT * D3 * sizeof(float)); // [B,3D,L]
    float* hb = pb;                                                // [B,L,4D] reuse

    // ---- embedding ----
    {
        size_t total = (size_t)MT * Dd;
        embed_kernel<<<(unsigned)((total + 255) / 256), 256, 0, stream>>>(
            ids, emb, u, MT, Dd);
    }

    const dim3 gemm_blk(256);
    const unsigned ln_grid = (MT + 7) / 8;

    for (int l = 0; l < NL; ++l) {
        // LN1: u -> z
        ln_kernel<<<ln_grid, 256, 0, stream>>>(u, ln1_g + (size_t)l * Dd,
                                               ln1_b + (size_t)l * Dd, zb, MT, Dd);

        // p = z @ Win + bin   [MT, 3D]
        gemm_wmma_kernel<<<dim3(D3 / TN, MT / TM), gemm_blk, 0, stream>>>(
            zb, Win + (size_t)l * Dd * D3, bin_ + (size_t)l * D3, pb,
            MT, D3, Dd, /*flags=*/0);

        // causal short conv + transpose: p[B,L,3D] -> q[B,3D,L]
        {
            size_t total = (size_t)Bn * D3 * Ln;
            shortconv_kernel<<<(unsigned)((total + 255) / 256), 256, 0, stream>>>(
                pb, convw + (size_t)l * D3 * 3, convb + (size_t)l * D3, qb,
                Bn, D3, Ln);
        }

        // fused gate + FFT long conv + bias + output gate -> y [B,L,D] (in zb)
        fftconv_kernel<<<Bn * Dd, 1024, (size_t)(2 * Ln) * sizeof(cplx), stream>>>(
            qb, hfil + (size_t)l * Dd * Ln, Db + (size_t)l * Dd, zb, Dd, Ln);

        // u += y @ Wout + bout
        gemm_wmma_kernel<<<dim3(Dd / TN, MT / TM), gemm_blk, 0, stream>>>(
            zb, Wout + (size_t)l * Dd * Dd, bout + (size_t)l * Dd, u,
            MT, Dd, Dd, /*flags=*/1);

        // LN2: u -> z
        ln_kernel<<<ln_grid, 256, 0, stream>>>(u, ln2_g + (size_t)l * Dd,
                                               ln2_b + (size_t)l * Dd, zb, MT, Dd);

        // hmid = gelu(z @ W1 + b1)  [MT, 4D]
        gemm_wmma_kernel<<<dim3(D4 / TN, MT / TM), gemm_blk, 0, stream>>>(
            zb, W1 + (size_t)l * Dd * D4, b1 + (size_t)l * D4, hb,
            MT, D4, Dd, /*flags=*/2);

        // u += hmid @ W2 + b2
        gemm_wmma_kernel<<<dim3(Dd / TN, MT / TM), gemm_blk, 0, stream>>>(
            hb, W2 + (size_t)l * D4 * Dd, b2 + (size_t)l * Dd, u,
            MT, Dd, D4, /*flags=*/1);
    }

    // final LN -> d_out
    ln_kernel<<<ln_grid, 256, 0, stream>>>(u, lnf_g, lnf_b, (float*)d_out, MT, Dd);
}
